// GCNConv_59150289601064
// MI455X (gfx1250) — matrix operand order repaired and checked by reading, not compiled, not run on hardware
//
#include <hip/hip_runtime.h>
#include <hip/hip_bf16.h>
#include <math.h>

#define NROWS 8192
#define INC   256
#define OUTC  256

typedef __attribute__((ext_vector_type(16))) __bf16 v16bf;
typedef __attribute__((ext_vector_type(8)))  __bf16 v8bf;
typedef __attribute__((ext_vector_type(8)))  float  v8f;
typedef __attribute__((ext_vector_type(4)))  unsigned int u32x4;
typedef __attribute__((ext_vector_type(8)))  int i32x8;
typedef __attribute__((ext_vector_type(4)))  int i32x4;

// TDM-staged adj tile: 64 rows x 32 K f32, padded to 36 dwords/row so the
// 16-row x float4 A-fragment reads are LDS bank-conflict free (36*r mod 64
// distinct for r=0..15).
#define TILE_K     32
#define TILE_M     64
#define LDS_ROW    36
#define TILE_DW    (TILE_M * LDS_ROW)       // 2304 dwords
#define TILE_BYTES (TILE_DW * 4)            // 9216 bytes

// ---------------------------------------------------------------------------
// Kernel 1: deg[i] = 1 + sum_j adj[i][j];  d_isqrt[i] = rsqrt(deg[i])
// ---------------------------------------------------------------------------
__global__ __launch_bounds__(256) void k_degree(const float* __restrict__ adj,
                                                float* __restrict__ d_isqrt) {
  __shared__ float red[256];
  const int row = blockIdx.x;
  const float* r = adj + (size_t)row * NROWS;
  float s = 0.f;
  for (int j = threadIdx.x; j < NROWS; j += 256) s += r[j];
  red[threadIdx.x] = s;
  __syncthreads();
  for (int off = 128; off > 0; off >>= 1) {
    if (threadIdx.x < off) red[threadIdx.x] += red[threadIdx.x + off];
    __syncthreads();
  }
  if (threadIdx.x == 0) d_isqrt[row] = rsqrtf(red[0] + 1.0f);
}

// ---------------------------------------------------------------------------
// Kernel 2: xsT[n][k] = bf16(d_isqrt[k] * x[k][n])   (scaled + transposed)
// ---------------------------------------------------------------------------
__global__ __launch_bounds__(256) void k_make_xsT(const float* __restrict__ x,
                                                  const float* __restrict__ d_isqrt,
                                                  __bf16* __restrict__ xsT) {
  __shared__ float t[32][33];
  const int k0 = blockIdx.x * 32;
  const int n0 = blockIdx.y * 32;
  const int tx = threadIdx.x & 31;
  const int ty = threadIdx.x >> 5;
#pragma unroll
  for (int i = 0; i < 4; ++i) {
    const int k = k0 + ty + i * 8;
    t[ty + i * 8][tx] = d_isqrt[k] * x[(size_t)k * INC + n0 + tx];
  }
  __syncthreads();
#pragma unroll
  for (int i = 0; i < 4; ++i) {
    const int n = n0 + ty + i * 8;
    xsT[(size_t)n * NROWS + k0 + tx] = (__bf16)t[tx][ty + i * 8];
  }
}

// ---------------------------------------------------------------------------
// Kernel 3: wT[o][c] = bf16(w[c][o])
// ---------------------------------------------------------------------------
__global__ __launch_bounds__(256) void k_make_wT(const float* __restrict__ w,
                                                 __bf16* __restrict__ wT) {
  const int i = blockIdx.x * 256 + threadIdx.x;
  const int o = i >> 8;
  const int c = i & 255;
  wT[i] = (__bf16)w[c * OUTC + o];
}

// ---------------------------------------------------------------------------
// Fragment helpers (bf16 16x16x32 WMMA layouts, wave32)
// ---------------------------------------------------------------------------
__device__ inline v16bf pack_a_f32(const float4 f0, const float4 f1,
                                   const float4 f2, const float4 f3) {
  v16bf a;
  a[0]  = (__bf16)f0.x; a[1]  = (__bf16)f0.y; a[2]  = (__bf16)f0.z; a[3]  = (__bf16)f0.w;
  a[4]  = (__bf16)f1.x; a[5]  = (__bf16)f1.y; a[6]  = (__bf16)f1.z; a[7]  = (__bf16)f1.w;
  a[8]  = (__bf16)f2.x; a[9]  = (__bf16)f2.y; a[10] = (__bf16)f2.z; a[11] = (__bf16)f2.w;
  a[12] = (__bf16)f3.x; a[13] = (__bf16)f3.y; a[14] = (__bf16)f3.z; a[15] = (__bf16)f3.w;
  return a;
}

__device__ inline v16bf pack_b_bf16(const v8bf b0, const v8bf b1) {
  v16bf b;
#pragma unroll
  for (int t = 0; t < 8; ++t) { b[t] = b0[t]; b[t + 8] = b1[t]; }
  return b;
}

// ---------------------------------------------------------------------------
// TDM: DMA one 64x32 f32 adj tile (row stride 8192) into LDS at ldsOff,
// padding LDS rows to 36 dwords (pad 4 dwords every 32 dwords).
// D# group0: count=1 | lds_addr | global_addr(57b) | type=2
// D# group1: mask=0, data_size=4B, pad_enable, interval=32dw(code 4),
//            amount=4dw(code 3), tensor_dim0/1=8192, tile_dim0=32,
//            tile_dim1=64, tensor_dim0_stride=8192
// ---------------------------------------------------------------------------
__device__ inline void tdm_load_adj_tile(const float* gptr, unsigned ldsOff) {
  const unsigned long long ga = (unsigned long long)(uintptr_t)gptr;
  u32x4 g0;
  g0[0] = 1u;                                   // count=1 (valid user D#)
  g0[1] = ldsOff;                               // LDS byte address
  g0[2] = (unsigned)(ga & 0xffffffffu);         // global_addr[31:0]
  g0[3] = (unsigned)(ga >> 32) | (2u << 30);    // global_addr[56:32] | type=2
  i32x8 g1;
  g1[0] = (2 << 16) | (1 << 20) | (4 << 22) | (3 << 25);
  g1[1] = (int)(8192u << 16);                   // tensor_dim0[15:0]
  g1[2] = (int)(8192u << 16);                   // tensor_dim0 hi | tensor_dim1 lo
  g1[3] = (int)(32u << 16);                     // tensor_dim1 hi | tile_dim0=32
  g1[4] = 64;                                   // tile_dim1=64, tile_dim2=0
  g1[5] = 8192;                                 // tensor_dim0_stride[31:0]
  g1[6] = 0;
  g1[7] = 0;
  i32x4 z4 = {0, 0, 0, 0};
#if __clang_major__ >= 23
  i32x8 z8 = {0, 0, 0, 0, 0, 0, 0, 0};
  __builtin_amdgcn_tensor_load_to_lds(g0, g1, z4, z4, z8, 0);
#else
  __builtin_amdgcn_tensor_load_to_lds(g0, g1, z4, z4, 0);
#endif
}

// ---------------------------------------------------------------------------
// Kernel 4: main aggregation GEMM, TDM double-buffered A tiles.
//   H = adj(f32->bf16 in regs) @ xs ; h = d_isqrt[m]*(H + xs_identity)
// block tile 64M x 256N, 8 waves (2M x 4N), wave tile 32x64 (8 wmma tiles)
// grid: NROWS/64 = 128 blocks; dynamic LDS = 2 * 9216 B
// ---------------------------------------------------------------------------
__global__ __launch_bounds__(256) void k_gemm_agg(const float* __restrict__ adj,
                                                  const __bf16* __restrict__ xsT,
                                                  const float* __restrict__ d_isqrt,
                                                  __bf16* __restrict__ hbuf) {
  extern __shared__ float lds[];              // [2][TILE_DW]
  const int lane = threadIdx.x & 31;
  const int wave = threadIdx.x >> 5;          // 0..7
  const int wm   = wave & 1;                  // 2 M-waves
  const int wn   = wave >> 1;                 // 4 N-waves
  const int l15  = lane & 15;
  const int half = lane >> 4;
  const int mBlock = blockIdx.x * TILE_M;

  v8f acc[2][4] = {};

  const float* adjBase = adj + (size_t)mBlock * NROWS;  // tile (mBlock, k0) = adjBase + k0

  const __bf16* bCol[4];
#pragma unroll
  for (int j = 0; j < 4; ++j)
    bCol[j] = xsT + (size_t)(wn * 64 + j * 16 + l15) * NROWS;

  if (wave == 0) tdm_load_adj_tile(adjBase, 0u);

  for (int k0 = 0; k0 < NROWS; k0 += TILE_K) {
    const int cur = (k0 >> 5) & 1;
    if (wave == 0) __builtin_amdgcn_s_wait_tensorcnt(0);
    __syncthreads();  // tile[cur] ready for all; prior reads of tile[cur^1] done
    if (wave == 0 && (k0 + TILE_K) < NROWS)
      tdm_load_adj_tile(adjBase + k0 + TILE_K,
                        (unsigned)((cur ^ 1) * TILE_BYTES));

    const float* tile = lds + cur * TILE_DW;
    v16bf a[2];
#pragma unroll
    for (int i = 0; i < 2; ++i) {
      const float* rp = tile + (wm * 32 + i * 16 + l15) * LDS_ROW + half * 8;
      const float4 f0 = *(const float4*)(rp);
      const float4 f1 = *(const float4*)(rp + 4);
      const float4 f2 = *(const float4*)(rp + 16);
      const float4 f3 = *(const float4*)(rp + 20);
      a[i] = pack_a_f32(f0, f1, f2, f3);
    }

    const int kB = k0 + half * 8;
#pragma unroll
    for (int j = 0; j < 4; ++j) {
      const v8bf b0 = *(const v8bf*)(bCol[j] + kB);
      const v8bf b1 = *(const v8bf*)(bCol[j] + kB + 16);
      const v16bf b = pack_b_bf16(b0, b1);
#pragma unroll
      for (int i = 0; i < 2; ++i)
        acc[i][j] = __builtin_amdgcn_wmma_f32_16x16x32_bf16(
            false, a[i], false, b, (short)0, acc[i][j], false, false);
    }
  }

  // epilogue: h[m][n] = d_isqrt[m] * (acc + xs[m][n]),  xs[m][n] == xsT[n][m]
#pragma unroll
  for (int i = 0; i < 2; ++i) {
#pragma unroll
    for (int j = 0; j < 4; ++j) {
      const int n = wn * 64 + j * 16 + l15;
#pragma unroll
      for (int r = 0; r < 8; ++r) {
        const int m = mBlock + wm * 32 + i * 16 + half * 8 + r;
        const float idt = (float)xsT[(size_t)n * NROWS + m];
        const float hv  = d_isqrt[m] * (acc[i][j][r] + idt);
        hbuf[(size_t)m * INC + n] = (__bf16)hv;
      }
    }
  }
}

// ---------------------------------------------------------------------------
// Kernel 5: out = ELU(h @ w)   (M=8192, N=256, K=256)
// ---------------------------------------------------------------------------
__global__ __launch_bounds__(256) void k_gemm_out(const __bf16* __restrict__ hbuf,
                                                  const __bf16* __restrict__ wT,
                                                  float* __restrict__ out) {
  const int lane = threadIdx.x & 31;
  const int wave = threadIdx.x >> 5;
  const int wm   = wave & 1;
  const int wn   = wave >> 1;
  const int l15  = lane & 15;
  const int half = lane >> 4;
  const int mBlock = blockIdx.x * 64;

  v8f acc[2][4] = {};

  const __bf16* aRow[2];
#pragma unroll
  for (int i = 0; i < 2; ++i)
    aRow[i] = hbuf + (size_t)(mBlock + wm * 32 + i * 16 + l15) * INC;

  const __bf16* bCol[4];
#pragma unroll
  for (int j = 0; j < 4; ++j)
    bCol[j] = wT + (size_t)(wn * 64 + j * 16 + l15) * INC;

#pragma unroll
  for (int k0 = 0; k0 < INC; k0 += 32) {
    v16bf a[2];
#pragma unroll
    for (int i = 0; i < 2; ++i) {
      const v8bf a0 = *(const v8bf*)(aRow[i] + k0 + half * 8);
      const v8bf a1 = *(const v8bf*)(aRow[i] + k0 + 16 + half * 8);
      a[i] = pack_b_bf16(a0, a1);
    }
    const int kB = k0 + half * 8;
#pragma unroll
    for (int j = 0; j < 4; ++j) {
      const v8bf b0 = *(const v8bf*)(bCol[j] + kB);
      const v8bf b1 = *(const v8bf*)(bCol[j] + kB + 16);
      const v16bf b = pack_b_bf16(b0, b1);
#pragma unroll
      for (int i = 0; i < 2; ++i)
        acc[i][j] = __builtin_amdgcn_wmma_f32_16x16x32_bf16(
            false, a[i], false, b, (short)0, acc[i][j], false, false);
    }
  }

#pragma unroll
  for (int i = 0; i < 2; ++i) {
#pragma unroll
    for (int j = 0; j < 4; ++j) {
      const int n = wn * 64 + j * 16 + l15;
#pragma unroll
      for (int r = 0; r < 8; ++r) {
        const int m = mBlock + wm * 32 + i * 16 + half * 8 + r;
        const float v = acc[i][j][r];
        out[(size_t)m * OUTC + n] = (v > 0.f) ? v : expm1f(v);
      }
    }
  }
}

// ---------------------------------------------------------------------------
// Host launcher. inputs: x[8192,256] f32, adj[8192,8192] f32, w[256,256] f32.
// Workspace (~8.3 MB): d_isqrt @0 | xsT bf16 @64K | wT bf16 @64K+4M |
//                      hbuf bf16 @64K+4M+256K
// ---------------------------------------------------------------------------
extern "C" void kernel_launch(void* const* d_in, const int* in_sizes, int n_in,
                              void* d_out, int out_size, void* d_ws, size_t ws_size,
                              hipStream_t stream) {
  const float* x   = (const float*)d_in[0];
  const float* adj = (const float*)d_in[1];
  const float* w   = (const float*)d_in[2];
  float* out = (float*)d_out;

  char* ws = (char*)d_ws;
  float*  d_isqrt = (float*)ws;
  __bf16* xsT  = (__bf16*)(ws + (64u << 10));
  __bf16* wT   = (__bf16*)(ws + (64u << 10) + (4u << 20));
  __bf16* hbuf = (__bf16*)(ws + (64u << 10) + (4u << 20) + (256u << 10));

  k_degree  <<<NROWS, 256, 0, stream>>>(adj, d_isqrt);
  k_make_xsT<<<dim3(NROWS / 32, INC / 32), 256, 0, stream>>>(x, d_isqrt, xsT);
  k_make_wT <<<(INC * OUTC) / 256, 256, 0, stream>>>(w, wT);
  k_gemm_agg<<<NROWS / 64, 256, 2 * TILE_BYTES, stream>>>(adj, xsT, d_isqrt, hbuf);
  k_gemm_out<<<NROWS / 64, 256, 0, stream>>>(hbuf, wT, out);
}
